// Extractor_20409684590833
// MI455X (gfx1250) — compile-verified, matrix operand order checked by reading, not run on hardware
//
#include <hip/hip_runtime.h>

typedef _Float16 half_t;
typedef __attribute__((ext_vector_type(16))) _Float16 v16h;
typedef __attribute__((ext_vector_type(8)))  _Float16 v8h;
typedef __attribute__((ext_vector_type(8)))  float    v8f;

union U16 { v16h v; v8h h[2]; };

struct Frag { v8h lo, hi; };

static __device__ inline Frag ldfrag(const half_t* p) {
  Frag f;
  f.lo = *(const v8h*)p;
  f.hi = *(const v8h*)(p + 16);
  return f;
}
static __device__ inline v16h cat(const Frag& f) {
  U16 u; u.h[0] = f.lo; u.h[1] = f.hi; return u.v;
}

// ---------------------------------------------------------------------------
// Generic WMMA GEMM:  C[m,n] = scale * sum_k A[m, acol+k] * B[brow+n, bcol+k]
//                     (+ bias[brow+n]) (+resid) (relu), batched over blockIdx.z
// A, B are f16 row-major; C written as f32 and/or f16.
// Block = 256 threads (8 waves). Block tile: 128 rows x 64 cols.
// Wave tile: 16 rows x 64 cols = 4x v_wmma_f32_16x16x32_f16 per K-step.
// Fragment loads are double-buffered (cur/nxt) to overlap VMEM with WMMA.
// ---------------------------------------------------------------------------
struct GemmP {
  const half_t* A; long abatch; int lda; int acol0; int acolz;
  const half_t* B; long bbatch; int ldb; int brow0; int browz; int bcol0; int bcolz;
  float* Cf; half_t* Ch; long cbatch; int ldc; int ccol0; int ccolz;
  const float* bias; const float* resid;
  float scale; int relu; int Kd;
};

__global__ void __launch_bounds__(256) gemm_wmma(GemmP p) {
  const int lane = threadIdx.x & 31;
  const int wave = threadIdx.x >> 5;
  const int z    = blockIdx.z;
  const int row0 = blockIdx.y * 128 + wave * 16;
  const int n0   = blockIdx.x * 64;
  const int r    = lane & 15;
  const int hlf  = lane >> 4;

  const half_t* A = p.A + (long)z * p.abatch;
  const half_t* B = p.B + (long)z * p.bbatch;
  const int acol = p.acol0 + z * p.acolz;
  const int bcol = p.bcol0 + z * p.bcolz;
  const int brow = p.brow0 + z * p.browz;

  // A fragment (16x32 f16): lane r holds row row0+r; halves 0..7 = K {hlf*8..+8},
  // halves 8..15 = K {16+hlf*8..+8}.  B (32x16) is the mirror layout per column.
  const half_t* aP = A + (long)(row0 + r) * p.lda + acol + hlf * 8;
  const half_t* bP[4];
#pragma unroll
  for (int t = 0; t < 4; ++t)
    bP[t] = B + (long)(brow + n0 + t * 16 + r) * p.ldb + bcol + hlf * 8;

  const v8f zero8 = {0.f, 0.f, 0.f, 0.f, 0.f, 0.f, 0.f, 0.f};
  v8f acc[4];
#pragma unroll
  for (int t = 0; t < 4; ++t) acc[t] = zero8;

  Frag aCur = ldfrag(aP);
  Frag bCur[4];
#pragma unroll
  for (int t = 0; t < 4; ++t) bCur[t] = ldfrag(bP[t]);

  int k0 = 0;
  for (; k0 + 32 < p.Kd; k0 += 32) {
    // prefetch next K-step into separate registers
    Frag aNxt = ldfrag(aP + k0 + 32);
    Frag bNxt[4];
#pragma unroll
    for (int t = 0; t < 4; ++t) bNxt[t] = ldfrag(bP[t] + k0 + 32);
    // consume current fragments
#pragma unroll
    for (int t = 0; t < 4; ++t)
      acc[t] = __builtin_amdgcn_wmma_f32_16x16x32_f16(
          false, cat(aCur), false, cat(bCur[t]), (short)0, acc[t], false, false);
    aCur = aNxt;
#pragma unroll
    for (int t = 0; t < 4; ++t) bCur[t] = bNxt[t];
  }
#pragma unroll
  for (int t = 0; t < 4; ++t)
    acc[t] = __builtin_amdgcn_wmma_f32_16x16x32_f16(
        false, cat(aCur), false, cat(bCur[t]), (short)0, acc[t], false, false);

  // Epilogue. C/D layout: lane<16 -> (M=g, N=lane); lane>=16 -> (M=8+g, N=lane-16)
  float* Cf  = p.Cf ? p.Cf + (long)z * p.cbatch : nullptr;
  half_t* Ch = p.Ch ? p.Ch + (long)z * p.cbatch : nullptr;
  const int ccolbase = p.ccol0 + z * p.ccolz;
#pragma unroll
  for (int t = 0; t < 4; ++t) {
    const int nl = n0 + t * 16 + r;
    const float bv = p.bias ? p.bias[brow + nl] : 0.f;
#pragma unroll
    for (int g = 0; g < 8; ++g) {
      const int m = row0 + hlf * 8 + g;
      float v = acc[t][g] * p.scale + bv;
      if (p.relu) v = fmaxf(v, 0.f);
      const long off = (long)m * p.ldc + ccolbase + nl;
      if (p.resid) v += p.resid[off];
      if (Cf) Cf[off] = v;
      if (Ch) Ch[off] = (half_t)v;
    }
  }
}

// ---------------------------------------------------------------------------
// phi2 + sum-pool fused: one block = one polynomial (64 monomial rows).
// S[p, n] = sum_m relu( X1[p*64+m, :] . Wphi2[n, :] + bias[n] )
// Block 128 threads (4 waves); wave w does rows 16w..16w+15, 64 cols.
// ---------------------------------------------------------------------------
__global__ void __launch_bounds__(128) gemm_pool(const half_t* __restrict__ A,
                                                 const half_t* __restrict__ B,
                                                 const float* __restrict__ bias,
                                                 half_t* __restrict__ Sh, int p0) {
  const int lane = threadIdx.x & 31;
  const int wave = threadIdx.x >> 5;
  const int row0 = blockIdx.y * 64 + wave * 16;  // row within chunk
  const int n0   = blockIdx.x * 64;
  const int r    = lane & 15;
  const int hlf  = lane >> 4;

  const half_t* aP = A + (long)(row0 + r) * 1024 + hlf * 8;
  const half_t* bP[4];
#pragma unroll
  for (int t = 0; t < 4; ++t)
    bP[t] = B + (long)(n0 + t * 16 + r) * 1024 + hlf * 8;

  const v8f zero8 = {0.f, 0.f, 0.f, 0.f, 0.f, 0.f, 0.f, 0.f};
  v8f acc[4];
#pragma unroll
  for (int t = 0; t < 4; ++t) acc[t] = zero8;

  Frag aCur = ldfrag(aP);
  Frag bCur[4];
#pragma unroll
  for (int t = 0; t < 4; ++t) bCur[t] = ldfrag(bP[t]);

  int k0 = 0;
  for (; k0 + 32 < 1024; k0 += 32) {
    Frag aNxt = ldfrag(aP + k0 + 32);
    Frag bNxt[4];
#pragma unroll
    for (int t = 0; t < 4; ++t) bNxt[t] = ldfrag(bP[t] + k0 + 32);
#pragma unroll
    for (int t = 0; t < 4; ++t)
      acc[t] = __builtin_amdgcn_wmma_f32_16x16x32_f16(
          false, cat(aCur), false, cat(bCur[t]), (short)0, acc[t], false, false);
    aCur = aNxt;
#pragma unroll
    for (int t = 0; t < 4; ++t) bCur[t] = bNxt[t];
  }
#pragma unroll
  for (int t = 0; t < 4; ++t)
    acc[t] = __builtin_amdgcn_wmma_f32_16x16x32_f16(
        false, cat(aCur), false, cat(bCur[t]), (short)0, acc[t], false, false);

  __shared__ float part[4][4][32];
#pragma unroll
  for (int t = 0; t < 4; ++t) {
    const float bv = bias[n0 + t * 16 + r];
    float s = 0.f;
#pragma unroll
    for (int g = 0; g < 8; ++g) s += fmaxf(acc[t][g] + bv, 0.f);
    part[wave][t][lane] = s;
  }
  __syncthreads();
  if (threadIdx.x < 64) {
    const int t = threadIdx.x >> 4, c = threadIdx.x & 15;
    float s = 0.f;
#pragma unroll
    for (int w = 0; w < 4; ++w) s += part[w][t][c] + part[w][t][c + 16];
    Sh[(long)(p0 + blockIdx.y) * 1024 + n0 + t * 16 + c] = (half_t)s;
  }
}

// ------------------------------ small kernels ------------------------------
__global__ void cvt_f32_f16(const float* __restrict__ s, half_t* __restrict__ d, long n) {
  long i = (long)blockIdx.x * blockDim.x + threadIdx.x;
  if (i < n) d[i] = (half_t)s[i];
}

// rows x 16 f32 -> rows x 32 f16 (zero padded K)
__global__ void pad16to32(const float* __restrict__ s, half_t* __restrict__ d, long rows) {
  long i = (long)blockIdx.x * blockDim.x + threadIdx.x;
  if (i >= rows * 32) return;
  long r = i >> 5; int c = (int)(i & 31);
  d[i] = (c < 16) ? (half_t)s[r * 16 + c] : (half_t)0.f;
}

// V [1024 x 512] -> Vt [512 x 1024]
__global__ void transpose_v(const half_t* __restrict__ V, half_t* __restrict__ Vt) {
  int i = blockIdx.x * blockDim.x + threadIdx.x;
  if (i >= 512 * 1024) return;
  int d = i >> 10, t = i & 1023;
  Vt[i] = V[t * 512 + d];
}

__global__ void __launch_bounds__(256) ln_kernel(const float* __restrict__ R,
                                                 const float* __restrict__ w,
                                                 const float* __restrict__ b,
                                                 float* __restrict__ Xf,
                                                 half_t* __restrict__ Xh) {
  const int row = blockIdx.x, tid = threadIdx.x;
  const float* rp = R + (long)row * 512;
  float x0 = rp[tid], x1 = rp[tid + 256];
  __shared__ float red[256];
  red[tid] = x0 + x1;
  __syncthreads();
  for (int st = 128; st > 0; st >>= 1) {
    if (tid < st) red[tid] += red[tid + st];
    __syncthreads();
  }
  const float mu = red[0] * (1.f / 512.f);
  __syncthreads();
  const float d0 = x0 - mu, d1 = x1 - mu;
  red[tid] = d0 * d0 + d1 * d1;
  __syncthreads();
  for (int st = 128; st > 0; st >>= 1) {
    if (tid < st) red[tid] += red[tid + st];
    __syncthreads();
  }
  const float inv = rsqrtf(red[0] * (1.f / 512.f) + 1e-5f);
  const float y0 = d0 * inv * w[tid] + b[tid];
  const float y1 = d1 * inv * w[tid + 256] + b[tid + 256];
  Xf[(long)row * 512 + tid] = y0;
  Xf[(long)row * 512 + tid + 256] = y1;
  Xh[(long)row * 512 + tid] = (half_t)y0;
  Xh[(long)row * 512 + tid + 256] = (half_t)y1;
}

// softmax over rows of 1024 f32 -> f16 out. grid = H*1024 rows.
__global__ void __launch_bounds__(256) softmax_kernel(const float* __restrict__ S,
                                                      half_t* __restrict__ A) {
  const long base = (long)blockIdx.x * 1024;
  const int tid = threadIdx.x;
  float v[4];
#pragma unroll
  for (int i = 0; i < 4; ++i) v[i] = S[base + tid + i * 256];
  __shared__ float red[256];
  red[tid] = fmaxf(fmaxf(v[0], v[1]), fmaxf(v[2], v[3]));
  __syncthreads();
  for (int st = 128; st > 0; st >>= 1) {
    if (tid < st) red[tid] = fmaxf(red[tid], red[tid + st]);
    __syncthreads();
  }
  const float m = red[0];
  __syncthreads();
  float e[4], s = 0.f;
#pragma unroll
  for (int i = 0; i < 4; ++i) { e[i] = __expf(v[i] - m); s += e[i]; }
  red[tid] = s;
  __syncthreads();
  for (int st = 128; st > 0; st >>= 1) {
    if (tid < st) red[tid] += red[tid + st];
    __syncthreads();
  }
  const float inv = 1.f / red[0];
#pragma unroll
  for (int i = 0; i < 4; ++i) A[base + tid + i * 256] = (half_t)(e[i] * inv);
}

__global__ void fill_neginf(float* __restrict__ out, long n) {
  long i = (long)blockIdx.x * blockDim.x + threadIdx.x;
  if (i < n) out[i] = -__builtin_inff();
}

__global__ void __launch_bounds__(128) scatter_dots(const float* __restrict__ X,
                                                    const int* __restrict__ rows,
                                                    const int* __restrict__ cols,
                                                    float* __restrict__ out) {
  const int k = blockIdx.x;
  const int r = rows[k], c = cols[k];
  __shared__ float red[128];
  float s = 0.f;
  for (int i = threadIdx.x; i < 512; i += 128)
    s += X[(long)r * 512 + i] * X[(long)c * 512 + i];
  red[threadIdx.x] = s;
  __syncthreads();
  for (int st = 64; st > 0; st >>= 1) {
    if (threadIdx.x < st) red[threadIdx.x] += red[threadIdx.x + st];
    __syncthreads();
  }
  if (threadIdx.x == 0) out[(long)r * 1024 + c] = red[0];
}

// ------------------------------- host side ---------------------------------
static inline void launch_gemm(hipStream_t st,
                               const half_t* A, long abatch, int lda, int acol0, int acolz,
                               const half_t* B, long bbatch, int ldb, int brow0, int browz,
                               int bcol0, int bcolz,
                               float* Cf, half_t* Ch, long cbatch, int ldc, int ccol0, int ccolz,
                               const float* bias, const float* resid, float scale, int relu,
                               int M, int N, int Kd, int Z) {
  GemmP p;
  p.A = A; p.abatch = abatch; p.lda = lda; p.acol0 = acol0; p.acolz = acolz;
  p.B = B; p.bbatch = bbatch; p.ldb = ldb; p.brow0 = brow0; p.browz = browz;
  p.bcol0 = bcol0; p.bcolz = bcolz;
  p.Cf = Cf; p.Ch = Ch; p.cbatch = cbatch; p.ldc = ldc; p.ccol0 = ccol0; p.ccolz = ccolz;
  p.bias = bias; p.resid = resid; p.scale = scale; p.relu = relu; p.Kd = Kd;
  dim3 grid(N / 64, M / 128, Z);
  gemm_wmma<<<grid, 256, 0, st>>>(p);
}

extern "C" void kernel_launch(void* const* d_in, const int* in_sizes, int n_in,
                              void* d_out, int out_size, void* d_ws, size_t ws_size,
                              hipStream_t stream) {
  (void)in_sizes; (void)n_in; (void)out_size; (void)ws_size;
  const int P = 1024, D = 512, HID = 1024, FF = 2048, L = 4, Kpairs = 2048;
  const int NROWS = 65536;        // P * M
  const int CHUNK = 8192;         // rows per phi-chain chunk (128 polys)

  const float* ideal   = (const float*)d_in[0];
  const int*   rowsI   = (const int*)d_in[1];
  const int*   colsI   = (const int*)d_in[2];
  const float* Wm      = (const float*)d_in[3];
  const float* bm      = (const float*)d_in[4];
  const float* W_phi1  = (const float*)d_in[5];
  const float* b_phi1  = (const float*)d_in[6];
  const float* W_phi2  = (const float*)d_in[7];
  const float* b_phi2  = (const float*)d_in[8];
  const float* W_rho1  = (const float*)d_in[9];
  const float* b_rho1  = (const float*)d_in[10];
  const float* W_rho2  = (const float*)d_in[11];
  const float* b_rho2  = (const float*)d_in[12];
  const float* Wq      = (const float*)d_in[13];
  const float* bq      = (const float*)d_in[14];
  const float* Wk      = (const float*)d_in[15];
  const float* bk      = (const float*)d_in[16];
  const float* Wv      = (const float*)d_in[17];
  const float* bv      = (const float*)d_in[18];
  const float* Wo      = (const float*)d_in[19];
  const float* bo      = (const float*)d_in[20];
  const float* ln1_w   = (const float*)d_in[21];
  const float* ln1_b   = (const float*)d_in[22];
  const float* W1      = (const float*)d_in[23];
  const float* b1      = (const float*)d_in[24];
  const float* W2      = (const float*)d_in[25];
  const float* b2      = (const float*)d_in[26];
  const float* ln2_w   = (const float*)d_in[27];
  const float* ln2_b   = (const float*)d_in[28];
  float* out = (float*)d_out;

  // ---- workspace bump allocator ----
  char* base = (char*)d_ws;
  auto alloc = [&](size_t bytes) -> char* {
    char* r = base; base += (bytes + 255) & ~(size_t)255; return r;
  };
  // f16 weights
  half_t* Wmh     = (half_t*)alloc((size_t)512 * 32 * 2);      // padded K 16->32
  half_t* Wphi1h  = (half_t*)alloc((size_t)HID * D * 2);
  half_t* Wphi2h  = (half_t*)alloc((size_t)HID * HID * 2);
  half_t* Wrho1h  = (half_t*)alloc((size_t)HID * HID * 2);
  half_t* Wrho2h  = (half_t*)alloc((size_t)D * HID * 2);
  half_t* Wqh     = (half_t*)alloc((size_t)L * D * D * 2);
  half_t* Wkh     = (half_t*)alloc((size_t)L * D * D * 2);
  half_t* Wvh     = (half_t*)alloc((size_t)L * D * D * 2);
  half_t* Woh     = (half_t*)alloc((size_t)L * D * D * 2);
  half_t* W1h     = (half_t*)alloc((size_t)L * FF * D * 2);
  half_t* W2h     = (half_t*)alloc((size_t)L * D * FF * 2);
  // activations
  half_t* idealh  = (half_t*)alloc((size_t)NROWS * 32 * 2);    // padded K
  half_t* X0h     = (half_t*)alloc((size_t)CHUNK * D * 2);
  half_t* X1h     = (half_t*)alloc((size_t)CHUNK * HID * 2);
  half_t* Sh      = (half_t*)alloc((size_t)P * HID * 2);
  half_t* S1h     = (half_t*)alloc((size_t)P * HID * 2);
  float*  Xf      = (float*)alloc((size_t)P * D * 4);
  half_t* Xh      = (half_t*)alloc((size_t)P * D * 2);
  float*  Rres    = (float*)alloc((size_t)P * D * 4);
  half_t* Qh      = (half_t*)alloc((size_t)P * D * 2);
  half_t* Kh      = (half_t*)alloc((size_t)P * D * 2);
  half_t* Vh      = (half_t*)alloc((size_t)P * D * 2);
  half_t* Vth     = (half_t*)alloc((size_t)P * D * 2);
  half_t* Oh      = (half_t*)alloc((size_t)P * D * 2);
  float*  scores  = (float*)alloc((size_t)8 * P * P * 4);
  half_t* attnh   = (half_t*)alloc((size_t)8 * P * P * 2);
  half_t* M1h     = (half_t*)alloc((size_t)P * FF * 2);

  auto cvt = [&](const float* s, half_t* d, long n) {
    cvt_f32_f16<<<(unsigned)((n + 255) / 256), 256, 0, stream>>>(s, d, n);
  };
  // ---- weight conversion (every call; deterministic) ----
  pad16to32<<<(512 * 32 + 255) / 256, 256, 0, stream>>>(Wm, Wmh, 512);
  pad16to32<<<(NROWS * 32 + 255) / 256, 256, 0, stream>>>(ideal, idealh, NROWS);
  cvt(W_phi1, Wphi1h, (long)HID * D);
  cvt(W_phi2, Wphi2h, (long)HID * HID);
  cvt(W_rho1, Wrho1h, (long)HID * HID);
  cvt(W_rho2, Wrho2h, (long)D * HID);
  cvt(Wq, Wqh, (long)L * D * D);  cvt(Wk, Wkh, (long)L * D * D);
  cvt(Wv, Wvh, (long)L * D * D);  cvt(Wo, Woh, (long)L * D * D);
  cvt(W1, W1h, (long)L * FF * D); cvt(W2, W2h, (long)L * D * FF);

  // ---- Stage A: monomial embed + DeepSets phi + pool (chunked over rows) ----
  for (int c = 0; c < NROWS / CHUNK; ++c) {
    const half_t* idc = idealh + (long)c * CHUNK * 32;
    launch_gemm(stream, idc, 0, 32, 0, 0,  Wmh, 0, 32, 0, 0, 0, 0,
                nullptr, X0h, 0, D, 0, 0,  bm, nullptr, 1.f, 1, CHUNK, D, 32, 1);
    launch_gemm(stream, X0h, 0, D, 0, 0,  Wphi1h, 0, D, 0, 0, 0, 0,
                nullptr, X1h, 0, HID, 0, 0,  b_phi1, nullptr, 1.f, 1, CHUNK, HID, D, 1);
    dim3 pg(HID / 64, CHUNK / 64);
    gemm_pool<<<pg, 128, 0, stream>>>(X1h, Wphi2h, b_phi2, Sh, c * (CHUNK / 64));
  }
  // rho
  launch_gemm(stream, Sh, 0, HID, 0, 0,  Wrho1h, 0, HID, 0, 0, 0, 0,
              nullptr, S1h, 0, HID, 0, 0,  b_rho1, nullptr, 1.f, 1, P, HID, HID, 1);
  launch_gemm(stream, S1h, 0, HID, 0, 0,  Wrho2h, 0, HID, 0, 0, 0, 0,
              Xf, Xh, 0, D, 0, 0,  b_rho2, nullptr, 1.f, 1, P, D, HID, 1);

  // ---- Stage B: transformer layers ----
  for (int l = 0; l < L; ++l) {
    const long wOff = (long)l * D * D;
    launch_gemm(stream, Xh, 0, D, 0, 0,  Wqh + wOff, 0, D, 0, 0, 0, 0,
                nullptr, Qh, 0, D, 0, 0,  bq + l * D, nullptr, 1.f, 0, P, D, D, 1);
    launch_gemm(stream, Xh, 0, D, 0, 0,  Wkh + wOff, 0, D, 0, 0, 0, 0,
                nullptr, Kh, 0, D, 0, 0,  bk + l * D, nullptr, 1.f, 0, P, D, D, 1);
    launch_gemm(stream, Xh, 0, D, 0, 0,  Wvh + wOff, 0, D, 0, 0, 0, 0,
                nullptr, Vh, 0, D, 0, 0,  bv + l * D, nullptr, 1.f, 0, P, D, D, 1);
    transpose_v<<<(512 * 1024 + 255) / 256, 256, 0, stream>>>(Vh, Vth);
    // scores[h,q,k] = 1/8 * Q[q, h*64:].K[k, h*64:]
    launch_gemm(stream, Qh, 0, D, 0, 64,  Kh, 0, D, 0, 0, 0, 64,
                scores, nullptr, (long)P * P, P, 0, 0,
                nullptr, nullptr, 0.125f, 0, P, P, 64, 8);
    softmax_kernel<<<8 * P, 256, 0, stream>>>(scores, attnh);
    // O[q, h*64+d] = sum_k attn[h,q,k] * Vt[h*64+d, k]
    launch_gemm(stream, attnh, (long)P * P, P, 0, 0,  Vth, 0, P, 0, 64, 0, 0,
                nullptr, Oh, 0, D, 0, 64,  nullptr, nullptr, 1.f, 0, P, 64, P, 8);
    // R = X + O @ Wo^T + bo ; X = LN1(R)
    launch_gemm(stream, Oh, 0, D, 0, 0,  Woh + wOff, 0, D, 0, 0, 0, 0,
                Rres, nullptr, 0, D, 0, 0,  bo + l * D, Xf, 1.f, 0, P, D, D, 1);
    ln_kernel<<<P, 256, 0, stream>>>(Rres, ln1_w + l * D, ln1_b + l * D, Xf, Xh);
    // FFN
    launch_gemm(stream, Xh, 0, D, 0, 0,  W1h + (long)l * FF * D, 0, D, 0, 0, 0, 0,
                nullptr, M1h, 0, FF, 0, 0,  b1 + l * FF, nullptr, 1.f, 1, P, FF, D, 1);
    launch_gemm(stream, M1h, 0, FF, 0, 0,  W2h + (long)l * D * FF, 0, FF, 0, 0, 0, 0,
                Rres, nullptr, 0, D, 0, 0,  b2 + l * D, Xf, 1.f, 0, P, D, FF, 1);
    ln_kernel<<<P, 256, 0, stream>>>(Rres, ln2_w + l * D, ln2_b + l * D, Xf, Xh);
  }

  // ---- Stage C: masked x @ x^T — only K needed entries ----
  fill_neginf<<<(P * P + 255) / 256, 256, 0, stream>>>(out, (long)P * P);
  scatter_dots<<<Kpairs, 128, 0, stream>>>(Xf, rowsI, colsI, out);
}
